// DeformDownsampleBlock_6313601925213
// MI455X (gfx1250) — matrix operand order repaired and checked by reading, not compiled, not run on hardware
//
#include <hip/hip_runtime.h>
#include <hip/hip_bf16.h>

// ---------------------------------------------------------------------------
// DeformDownsampleBlock on MI455X (gfx1250, wave32, WMMA).
// maxpool residual | offset-conv (WMMA) | deform conv (gather + WMMA, fused
// BN1 stats, k-major K packing for 4x coordinate reuse) | BN1+ReLU->bf16 |
// conv2 (WMMA, fused BN2 stats) | BN2+res+ReLU.
// ---------------------------------------------------------------------------

typedef __attribute__((ext_vector_type(16))) __bf16 v16bf;
typedef __attribute__((ext_vector_type(8)))  __bf16 v8bf;
typedef __attribute__((ext_vector_type(8)))  float  v8f;

#define NB    4
#define CIN   64
#define DD    16
#define HH    56
#define WW    56
#define HWIN  (HH*WW)          // 3136
#define DHWIN (DD*HWIN)        // 50176
#define DOT   8
#define HOT   28
#define WOT   28
#define SPO   (HOT*WOT)        // 784
#define SP    (DOT*SPO)        // 6272
#define NP    (NB*SP)          // 25088
#define OFFC  648
#define KK1   (CIN*27)         // 1728
#define COUT  128
#define KK2   (COUT*27)        // 3456
#define NCNT  25088.0f
#define POSB  32               // positions per workgroup
#define BST   40               // Bsh row stride (halfwords), bank-conflict free
#define EPT   4                // fill elements per thread (POSB*32/256)

// packed taps: kz | ky<<2 | kx<<4  for k = kz*9+ky*3+kx
__constant__ int TAP3[27] = {
    0, 16, 32,  4, 20, 36,  8, 24, 40,
    1, 17, 33,  5, 21, 37,  9, 25, 41,
    2, 18, 34,  6, 22, 38, 10, 26, 42
};

static __device__ __forceinline__ int clampi(int v, int hi) {
    return min(max(v, 0), hi);           // v_med3_i32
}

// ---------------------------------------------------------------------------
__global__ __launch_bounds__(256) void cvt_bf16_kernel(const float* __restrict__ s,
                                                       __bf16* __restrict__ d, int n) {
    int i = blockIdx.x * 256 + threadIdx.x;
    if (i < n) d[i] = (__bf16)s[i];
}

// w1 repack to k-major K ordering: dst[oc][k*64+c] = src[oc][c*27+k]
__global__ __launch_bounds__(256) void repack_w1_kernel(const float* __restrict__ s,
                                                        __bf16* __restrict__ d) {
    int i = blockIdx.x * 256 + threadIdx.x;
    if (i >= COUT * KK1) return;
    int oc = i / KK1, r = i - oc * KK1;
    int c = r / 27, k = r - c * 27;
    d[oc * KK1 + k * CIN + c] = (__bf16)s[i];
}

// ---------------------------------------------------------------------------
// 3x3x3 maxpool, stride 2, pad 1
// ---------------------------------------------------------------------------
__global__ __launch_bounds__(256) void maxpool_kernel(const float* __restrict__ x,
                                                      float* __restrict__ resid) {
    int i = blockIdx.x * 256 + threadIdx.x;
    if (i >= NB * CIN * SP) return;
    int n = i / (CIN * SP);
    int c = (i / SP) % CIN;
    int sp = i % SP;
    int dz = sp / SPO; int r = sp % SPO;
    int hy = r / WOT;  int wx = r % WOT;
    const float* base = x + ((size_t)(n * CIN + c)) * DHWIN;
    float m = -INFINITY;
    #pragma unroll
    for (int kz = 0; kz < 3; ++kz) {
        int zi = dz * 2 - 1 + kz; if ((unsigned)zi >= DD) continue;
        #pragma unroll
        for (int ky = 0; ky < 3; ++ky) {
            int yi = hy * 2 - 1 + ky; if ((unsigned)yi >= HH) continue;
            #pragma unroll
            for (int kx = 0; kx < 3; ++kx) {
                int xi = wx * 2 - 1 + kx; if ((unsigned)xi >= WW) continue;
                m = fmaxf(m, base[(zi * HH + yi) * WW + xi]);
            }
        }
    }
    resid[i] = m;
}

// ---------------------------------------------------------------------------
// WMMA operand loads (CDNA5 ISA layouts), 16-byte pieces.
// ---------------------------------------------------------------------------
static __device__ __forceinline__ v16bf load_a_row(const __bf16* __restrict__ ap,
                                                   int laneHi) {
    v8bf a0 = *(const v8bf*)(ap + laneHi * 8);
    v8bf a1 = *(const v8bf*)(ap + laneHi * 8 + 16);
    v16bf a;
    #pragma unroll
    for (int i = 0; i < 8; ++i) { a[i] = a0[i]; a[i + 8] = a1[i]; }
    return a;
}

static __device__ __forceinline__ v16bf load_b_tile(const __bf16* bp) {
    v8bf b0 = *(const v8bf*)bp;
    v8bf b1 = *(const v8bf*)(bp + 16);
    v16bf b;
    #pragma unroll
    for (int i = 0; i < 8; ++i) { b[i] = b0[i]; b[i + 8] = b1[i]; }
    return b;
}

// incremental (c,k) state update for ck += 32 with k = ck % 27
static __device__ __forceinline__ void bump_ck(int& c, int& k) {
    k += 32;
    if (k >= 27) { k -= 27; ++c; }
    if (k >= 27) { k -= 27; ++c; }
}

// ---------------------------------------------------------------------------
// Offset conv: off[n,oc,p] = sum_{c,k} Woff[oc,c*27+k]*x[n,c,samp(p,k)] + b
// grid=(NP/32, 6), block 256.
// ---------------------------------------------------------------------------
__global__ __launch_bounds__(256) void off_gemm_kernel(const float* __restrict__ x,
                                                       const __bf16* __restrict__ wB,
                                                       const float* __restrict__ bOff,
                                                       float* __restrict__ off) {
    __shared__ __bf16 Bsh[POSB * BST];
    const int tid = threadIdx.x;
    const int lane = tid & 31;
    const int laneLo = lane & 15, laneHi = lane >> 4;
    const int p0 = blockIdx.x * POSB;
    const int n = p0 / SP, spb = p0 - n * SP;       // SP % POSB == 0: single n
    const int ocBase = blockIdx.y * 128 + (tid >> 5) * 16;
    const bool active = ocBase < OFFC;
    const int rowA = active ? min(ocBase + laneLo, OFFC - 1) : 0;
    const int xbase = n * CIN * DHWIN;

    const int kkf = tid & 31;
    int c = (kkf >= 27) ? 1 : 0;
    int k = kkf - 27 * c;
    int dzv[EPT], hyv[EPT], wxv[EPT];
    #pragma unroll
    for (int j = 0; j < EPT; ++j) {
        int sp = spb + (tid >> 5) + j * 8;
        dzv[j] = sp / SPO; int r = sp - dzv[j] * SPO;
        hyv[j] = r / WOT;  wxv[j] = r - hyv[j] * WOT;
    }

    const __bf16* ap = wB + (size_t)rowA * KK1;
    v8f acc0 = {}, acc1 = {};
    for (int kc = 0; kc < KK1; kc += 32) {
        __syncthreads();
        {
            const int tap = TAP3[k];
            const int kz = tap & 3, ky = (tap >> 2) & 3, kx = (tap >> 4) & 3;
            const int cb = xbase + c * DHWIN;
            float vals[EPT];
            bool  okv[EPT];
            #pragma unroll
            for (int j = 0; j < EPT; ++j) {       // phase 1: gathers in flight
                const int zi = dzv[j] * 2 - 1 + kz;
                const int yi = hyv[j] * 2 - 1 + ky;
                const int xi = wxv[j] * 2 - 1 + kx;
                okv[j] = (unsigned)zi < DD && (unsigned)yi < HH && (unsigned)xi < WW;
                const int idx = cb + clampi(zi, DD - 1) * HWIN +
                                clampi(yi, HH - 1) * WW + clampi(xi, WW - 1);
                vals[j] = x[idx];
            }
            #pragma unroll
            for (int j = 0; j < EPT; ++j)         // phase 2: convert + store
                Bsh[((tid >> 5) + j * 8) * BST + kkf] =
                    (__bf16)(okv[j] ? vals[j] : 0.f);
            bump_ck(c, k);
        }
        __syncthreads();
        if (active) {
            if (kc + 32 < KK1) __builtin_prefetch(ap + 32, 0, 1);
            v16bf a  = load_a_row(ap, laneHi);
            v16bf b0 = load_b_tile(&Bsh[(laneLo)      * BST + laneHi * 8]);
            v16bf b1 = load_b_tile(&Bsh[(16 + laneLo) * BST + laneHi * 8]);
            acc0 = __builtin_amdgcn_wmma_f32_16x16x32_bf16(false, a, false, b0,
                                                           (short)0, acc0, false, false);
            acc1 = __builtin_amdgcn_wmma_f32_16x16x32_bf16(false, a, false, b1,
                                                           (short)0, acc1, false, false);
        }
        ap += 32;
    }
    if (!active) return;
    #pragma unroll
    for (int r = 0; r < 8; ++r) {
        const int oc = ocBase + r + laneHi * 8;
        if (oc < OFFC) {
            const float bias = bOff[oc];
            const int base = (n * OFFC + oc) * SP + spb + laneLo;
            off[base]      = acc0[r] + bias;
            off[base + 16] = acc1[r] + bias;
        }
    }
}

// ---------------------------------------------------------------------------
// Deformable conv + fused BN1 stats.  K packed k-major (K = k*64 + c): a
// 32-wide K chunk has ONE wave-uniform tap k spanning 32 channels.  Each
// thread owns (position, 4-channel quad): offsets loaded once, trilinear
// (index, weight) pairs computed once, reused across its 4 channels.
// grid = NP/32, block 256.
// ---------------------------------------------------------------------------
__global__ __launch_bounds__(256) void deform_gemm_kernel(const float* __restrict__ x,
                                                          const float* __restrict__ off,
                                                          const __bf16* __restrict__ w1R,
                                                          float* __restrict__ raw1,
                                                          float* __restrict__ stats) {
    __shared__ __bf16 Bsh[POSB * BST];
    const int tid = threadIdx.x;
    const int lane = tid & 31;
    const int laneLo = lane & 15, laneHi = lane >> 4;
    const int p0 = blockIdx.x * POSB;
    const int n = p0 / SP, spb = p0 - n * SP;
    const int ocBase = (tid >> 5) * 16;
    const int xbase = n * CIN * DHWIN;
    const int obN = n * OFFC * SP;

    // fill-role decomposition: one position, one 4-channel quad per thread
    const int pos = tid >> 3;            // 0..31
    const int cq  = tid & 7;             // channel quad within 32-channel chunk
    const int sp  = spb + pos;
    const int dz  = sp / SPO; int rr = sp - dz * SPO;
    const int hy  = rr / WOT; const int wx = rr - hy * WOT;

    const __bf16* ap = w1R + (size_t)(ocBase + laneLo) * KK1;
    v8f acc0 = {}, acc1 = {};
    for (int k27 = 0; k27 < 27; ++k27) {
        const int tap = TAP3[k27];       // wave-uniform
        const int kz = tap & 3, ky = (tap >> 2) & 3, kx = (tap >> 4) & 3;
        #pragma unroll 1
        for (int half = 0; half < 2; ++half) {
            __syncthreads();
            {
                const int c0 = half * 32 + (cq << 2);
                const int g  = c0 >> 3;                       // cpg = 8
                const int ob = obN + (g * 81 + k27) * SP + sp;
                const float pz = (float)(dz * 2 - 1 + kz) + off[ob];
                const float py = (float)(hy * 2 - 1 + ky) + off[ob + 27 * SP];
                const float px = (float)(wx * 2 - 1 + kx) + off[ob + 54 * SP];
                const int z0 = (int)floorf(pz); const float fz = pz - (float)z0;
                const int y0 = (int)floorf(py); const float fy = py - (float)y0;
                const int x0 = (int)floorf(px); const float fx = px - (float)x0;
                int   idx[8];
                float wgt[8];
                #pragma unroll
                for (int t = 0; t < 8; ++t) {
                    const int dzt = t >> 2, dyt = (t >> 1) & 1, dxt = t & 1;
                    const int zi = z0 + dzt, yi = y0 + dyt, xi = x0 + dxt;
                    const bool ok = (unsigned)zi < DD && (unsigned)yi < HH &&
                                    (unsigned)xi < WW;
                    const float w = (dzt ? fz : 1.f - fz) * (dyt ? fy : 1.f - fy) *
                                    (dxt ? fx : 1.f - fx);
                    idx[t] = clampi(zi, DD - 1) * HWIN + clampi(yi, HH - 1) * WW +
                             clampi(xi, WW - 1);
                    wgt[t] = ok ? w : 0.f;
                }
                const float* xc = x + xbase + c0 * DHWIN;
                #pragma unroll
                for (int i = 0; i < 4; ++i) {
                    float a = 0.f;
                    #pragma unroll
                    for (int t = 0; t < 8; ++t) a += wgt[t] * xc[idx[t]];
                    Bsh[pos * BST + (cq << 2) + i] = (__bf16)a;
                    xc += DHWIN;
                }
            }
            __syncthreads();
            __builtin_prefetch(ap + 32, 0, 1);
            v16bf a  = load_a_row(ap, laneHi);
            v16bf b0 = load_b_tile(&Bsh[(laneLo)      * BST + laneHi * 8]);
            v16bf b1 = load_b_tile(&Bsh[(16 + laneLo) * BST + laneHi * 8]);
            acc0 = __builtin_amdgcn_wmma_f32_16x16x32_bf16(false, a, false, b0,
                                                           (short)0, acc0, false, false);
            acc1 = __builtin_amdgcn_wmma_f32_16x16x32_bf16(false, a, false, b1,
                                                           (short)0, acc1, false, false);
            ap += 32;
        }
    }
    #pragma unroll
    for (int r = 0; r < 8; ++r) {
        const int oc = ocBase + r + laneHi * 8;
        const int base = (n * COUT + oc) * SP + spb + laneLo;
        const float v0 = acc0[r], v1 = acc1[r];
        raw1[base]      = v0;
        raw1[base + 16] = v1;
        float s = v0 + v1, s2 = v0 * v0 + v1 * v1;
        #pragma unroll
        for (int m = 1; m < 16; m <<= 1) {
            s  += __shfl_xor(s,  m, 32);
            s2 += __shfl_xor(s2, m, 32);
        }
        if (laneLo == 0) {
            atomicAdd(&stats[oc], s);
            atomicAdd(&stats[COUT + oc], s2);
        }
    }
}

// ---------------------------------------------------------------------------
// BN1 (biased var) + ReLU -> bf16
// ---------------------------------------------------------------------------
__global__ __launch_bounds__(256) void bn1_relu_kernel(const float* __restrict__ raw1,
                                                       const float* __restrict__ stats,
                                                       const float* __restrict__ gam,
                                                       const float* __restrict__ bet,
                                                       __bf16* __restrict__ act1) {
    int i = blockIdx.x * 256 + threadIdx.x;
    if (i >= NB * COUT * SP) return;
    int c = (i / SP) % COUT;
    float mean = stats[c] * (1.f / NCNT);
    float var  = stats[COUT + c] * (1.f / NCNT) - mean * mean;
    float inv  = rsqrtf(var + 1e-5f);
    float v = gam[c] * (raw1[i] - mean) * inv + bet[c];
    act1[i] = (__bf16)fmaxf(v, 0.f);
}

// ---------------------------------------------------------------------------
// Conv2 (3x3x3 s1 p1) on act1(4,128,8,28,28), K=3456; fused BN2 stats.
// (conv bias b2 is absorbed by the following mean-subtracting BN.)
// ---------------------------------------------------------------------------
__global__ __launch_bounds__(256) void conv2_gemm_kernel(const __bf16* __restrict__ act1,
                                                         const __bf16* __restrict__ w2B,
                                                         float* __restrict__ raw2,
                                                         float* __restrict__ stats) {
    __shared__ __bf16 Bsh[POSB * BST];
    const int tid = threadIdx.x;
    const int lane = tid & 31;
    const int laneLo = lane & 15, laneHi = lane >> 4;
    const int p0 = blockIdx.x * POSB;
    const int n = p0 / SP, spb = p0 - n * SP;
    const int ocBase = (tid >> 5) * 16;
    const int abase = n * COUT * DOT * SPO;

    const int kkf = tid & 31;
    int c = (kkf >= 27) ? 1 : 0;
    int k = kkf - 27 * c;
    int dzv[EPT], hyv[EPT], wxv[EPT];
    #pragma unroll
    for (int j = 0; j < EPT; ++j) {
        int sp = spb + (tid >> 5) + j * 8;
        dzv[j] = sp / SPO; int r = sp - dzv[j] * SPO;
        hyv[j] = r / WOT;  wxv[j] = r - hyv[j] * WOT;
    }

    const __bf16* ap = w2B + (size_t)(ocBase + laneLo) * KK2;
    v8f acc0 = {}, acc1 = {};
    for (int kc = 0; kc < KK2; kc += 32) {
        __syncthreads();
        {
            const int tap = TAP3[k];
            const int kz = tap & 3, ky = (tap >> 2) & 3, kx = (tap >> 4) & 3;
            const int cb = abase + c * (DOT * SPO);
            __bf16 vals[EPT];
            bool   okv[EPT];
            #pragma unroll
            for (int j = 0; j < EPT; ++j) {       // phase 1: gathers in flight
                const int zi = dzv[j] - 1 + kz;
                const int yi = hyv[j] - 1 + ky;
                const int xi = wxv[j] - 1 + kx;
                okv[j] = (unsigned)zi < DOT && (unsigned)yi < HOT && (unsigned)xi < WOT;
                const int idx = cb + clampi(zi, DOT - 1) * SPO +
                                clampi(yi, HOT - 1) * WOT + clampi(xi, WOT - 1);
                vals[j] = act1[idx];
            }
            #pragma unroll
            for (int j = 0; j < EPT; ++j)         // phase 2: select + store
                Bsh[((tid >> 5) + j * 8) * BST + kkf] =
                    okv[j] ? vals[j] : (__bf16)0.f;
            bump_ck(c, k);
        }
        __syncthreads();
        if (kc + 32 < KK2) __builtin_prefetch(ap + 32, 0, 1);
        v16bf a  = load_a_row(ap, laneHi);
        v16bf b0 = load_b_tile(&Bsh[(laneLo)      * BST + laneHi * 8]);
        v16bf b1 = load_b_tile(&Bsh[(16 + laneLo) * BST + laneHi * 8]);
        acc0 = __builtin_amdgcn_wmma_f32_16x16x32_bf16(false, a, false, b0,
                                                       (short)0, acc0, false, false);
        acc1 = __builtin_amdgcn_wmma_f32_16x16x32_bf16(false, a, false, b1,
                                                       (short)0, acc1, false, false);
        ap += 32;
    }
    #pragma unroll
    for (int r = 0; r < 8; ++r) {
        const int oc = ocBase + r + laneHi * 8;
        const int base = (n * COUT + oc) * SP + spb + laneLo;
        const float v0 = acc0[r], v1 = acc1[r];
        raw2[base]      = v0;
        raw2[base + 16] = v1;
        float s = v0 + v1, s2 = v0 * v0 + v1 * v1;
        #pragma unroll
        for (int m = 1; m < 16; m <<= 1) {
            s  += __shfl_xor(s,  m, 32);
            s2 += __shfl_xor(s2, m, 32);
        }
        if (laneLo == 0) {
            atomicAdd(&stats[2 * COUT + oc], s);
            atomicAdd(&stats[3 * COUT + oc], s2);
        }
    }
}

// ---------------------------------------------------------------------------
// BN2 + duplicated-residual + ReLU -> d_out region 1
// ---------------------------------------------------------------------------
__global__ __launch_bounds__(256) void final_kernel(const float* __restrict__ raw2,
                                                    const float* __restrict__ stats,
                                                    const float* __restrict__ gam,
                                                    const float* __restrict__ bet,
                                                    const float* __restrict__ resid,
                                                    float* __restrict__ out) {
    int i = blockIdx.x * 256 + threadIdx.x;
    if (i >= NB * COUT * SP) return;
    int n = i / (COUT * SP);
    int c = (i / SP) % COUT;
    int sp = i % SP;
    float mean = stats[2 * COUT + c] * (1.f / NCNT);
    float var  = stats[3 * COUT + c] * (1.f / NCNT) - mean * mean;
    float inv  = rsqrtf(var + 1e-5f);
    float v = gam[c] * (raw2[i] - mean) * inv + bet[c];
    v += resid[(n * CIN + (c & 63)) * SP + sp];   // concat(residual, residual)
    out[i] = fmaxf(v, 0.f);
}

// ---------------------------------------------------------------------------
// Host launch
// ---------------------------------------------------------------------------
extern "C" void kernel_launch(void* const* d_in, const int* in_sizes, int n_in,
                              void* d_out, int out_size, void* d_ws, size_t ws_size,
                              hipStream_t stream) {
    const float* x      = (const float*)d_in[0];
    const float* w_off  = (const float*)d_in[1];
    const float* b_off  = (const float*)d_in[2];
    const float* w1     = (const float*)d_in[3];
    const float* bn1_g  = (const float*)d_in[4];
    const float* bn1_b  = (const float*)d_in[5];
    const float* w2     = (const float*)d_in[6];
    /* b2 (d_in[7]) is absorbed by BN2's mean subtraction */
    const float* bn2_g  = (const float*)d_in[8];
    const float* bn2_b  = (const float*)d_in[9];

    float* out_main = (float*)d_out;                       // (4,128,8,28,28)
    float* out_off  = (float*)d_out + NB * COUT * SP;      // (4,648,8,28,28)

    char* w = (char*)d_ws;
    float*  resid = (float*)(w + 0);                       //  6,422,528 B
    float*  raw1  = (float*)(w + 6422528);                 // 12,845,056 B
    __bf16* act1  = (__bf16*)(w + 19267584);               //  6,422,528 B
    float*  raw2  = (float*)(w + 25690112);                // 12,845,056 B
    __bf16* wOffB = (__bf16*)(w + 38535168);               //  2,239,488 B
    __bf16* w1R   = (__bf16*)(w + 40774656);               //    442,368 B
    __bf16* w2B   = (__bf16*)(w + 41217024);               //    884,736 B
    float*  stats = (float*)(w + 42101760);                //      2,048 B

    cvt_bf16_kernel<<<(OFFC * KK1 + 255) / 256, 256, 0, stream>>>(w_off, wOffB, OFFC * KK1);
    repack_w1_kernel<<<(COUT * KK1 + 255) / 256, 256, 0, stream>>>(w1, w1R);
    cvt_bf16_kernel<<<(COUT * KK2 + 255) / 256, 256, 0, stream>>>(w2, w2B, COUT * KK2);

    maxpool_kernel<<<(NB * CIN * SP + 255) / 256, 256, 0, stream>>>(x, resid);

    off_gemm_kernel<<<dim3(NP / POSB, 6), 256, 0, stream>>>(x, wOffB, b_off, out_off);

    hipMemsetAsync(stats, 0, 4 * COUT * sizeof(float), stream);

    deform_gemm_kernel<<<NP / POSB, 256, 0, stream>>>(x, out_off, w1R, raw1, stats);

    bn1_relu_kernel<<<(NB * COUT * SP + 255) / 256, 256, 0, stream>>>(raw1, stats,
                                                                      bn1_g, bn1_b, act1);

    conv2_gemm_kernel<<<NP / POSB, 256, 0, stream>>>(act1, w2B, raw2, stats);

    final_kernel<<<(NB * COUT * SP + 255) / 256, 256, 0, stream>>>(raw2, stats,
                                                                   bn2_g, bn2_b,
                                                                   resid, out_main);
    (void)in_sizes; (void)n_in; (void)out_size; (void)ws_size;
}
